// OuterProductMean_73254962201334
// MI455X (gfx1250) — matrix operand order repaired and checked by reading, not compiled, not run on hardware
//
#include <hip/hip_runtime.h>
#include <hip/hip_bf16.h>

// ---------------------------------------------------------------------------
// OuterProductMean for MI455X (gfx1250, wave32, WMMA).
// B=1, M=128, L=256, IN_C=256, MID_C=32, OUT_C=128.
//
// k1: LayerNorm + projection -> bf16 Ibuf[l][x][m], Jt[l][y][m]  (K=m inner)
// k2: per (2-li block, 32-lj block):
//      GEMM1  S[li][x, lj*32+y] = sum_m I[li][x,m] * J[lj*32+y, m]  (wmma bf16)
//      GEMM2  OUT[li][lj, o]    = sum_f S[li][lj,f] * Wf[o,f]       (wmma bf16)
// Blocking chosen to cut L2 traffic: each J B-fragment is reused for 2 li,
// each Wf B-fragment for 4 output tiles -> ~512MB total L2 reads, balanced
// against ~7us of dense-bf16 WMMA compute.
// ---------------------------------------------------------------------------

typedef __bf16 v16bf __attribute__((ext_vector_type(16)));
typedef __bf16 v8bf  __attribute__((ext_vector_type(8)));
typedef __bf16 v2bf  __attribute__((ext_vector_type(2)));
typedef float  v8f   __attribute__((ext_vector_type(8)));
typedef int    v4i   __attribute__((ext_vector_type(4)));

typedef __attribute__((address_space(1))) v4i* gptr_b128;
typedef __attribute__((address_space(3))) v4i* lptr_b128;

#if defined(__has_builtin)
#if __has_builtin(__builtin_amdgcn_global_load_async_to_lds_b128)
#define OPM_ASYNC_LDS 1
#endif
#endif

// Per-lane async global->LDS 16B copy (ASYNCcnt path), with sync fallback.
__device__ __forceinline__ void async_copy_b128(const __bf16* g, __bf16* l) {
#if defined(OPM_ASYNC_LDS)
  __builtin_amdgcn_global_load_async_to_lds_b128(
      (gptr_b128)g, (lptr_b128)l, /*offset=*/0, /*cpol=*/0);
#else
  *(uint4*)l = *(const uint4*)g;
#endif
}

__device__ __forceinline__ void wait_async_all() {
#if defined(OPM_ASYNC_LDS)
#if __has_builtin(__builtin_amdgcn_s_wait_asynccnt)
  __builtin_amdgcn_s_wait_asynccnt(0);
#else
  asm volatile("s_wait_asynccnt 0x0" ::: "memory");
#endif
#endif
}

// A-fragment (16x32 bf16, M x K) from a row-major row pointer.
// Layout: lanes 0-15 = rows M=0..15 with K = {kb+hi*16..+16} contiguous in
// elems 0..15; lanes 16-31 mirror with hi=1 (matches contiguous B loads).
__device__ __forceinline__ v16bf load_a_frag(const __bf16* row, int kb, int hi) {
  const v8bf lo = *(const v8bf*)(row + kb + hi * 16);
  const v8bf hh = *(const v8bf*)(row + kb + hi * 16 + 8);
  v16bf r;
#pragma unroll
  for (int i = 0; i < 8; ++i) { r[i] = lo[i]; r[8 + i] = hh[i]; }
  return r;
}

__device__ __forceinline__ v8f wmma_bf16(v16bf a, v16bf b, v8f c) {
  return __builtin_amdgcn_wmma_f32_16x16x32_bf16(
      /*neg_a=*/false, a, /*neg_b=*/false, b,
      /*c_mod=*/(short)0, c, /*reuse_a=*/false, /*reuse_b=*/false);
}

// ---------------------------------------------------------------------------
// Kernel 1: LayerNorm over 256 channels + Linear(256 -> 64), interleave-split.
// One wave per token (m,l); 8 tokens per 256-thread block.
//   Ibuf[(l*32 + x)*128 + m] = p[m,l,2x]     (even channels -> i)
//   Jt  [(l*32 + y)*128 + m] = p[m,l,2y+1]   (odd channels  -> j)
// ---------------------------------------------------------------------------
__global__ __launch_bounds__(256) void opm_ln_proj(
    const float* __restrict__ x1d, const float* __restrict__ norm_w,
    const float* __restrict__ norm_b, const float* __restrict__ proj_w,
    const float* __restrict__ proj_b, __bf16* __restrict__ Ibuf,
    __bf16* __restrict__ Jt) {
  __shared__ __bf16 Wp[256 * 64];    // transposed proj_w: Wp[c][d], bf16 (32KB)
  __shared__ float  xtok[8][256];    // normalized token per wave (8KB)

  const int tid = threadIdx.x;
  for (int i = tid; i < 64 * 256; i += 256) {
    const int d = i >> 8, c = i & 255;
    Wp[c * 64 + d] = (__bf16)proj_w[i];
  }
  __syncthreads();

  const int wave = tid >> 5, lane = tid & 31;
  const int t = blockIdx.x * 8 + wave;  // token id = m*256 + l
  const int m = t >> 8, l = t & 255;
  const float* xp = x1d + (size_t)t * 256;

  float xv[8], s1 = 0.f, s2 = 0.f;
#pragma unroll
  for (int k = 0; k < 8; ++k) {
    const float v = xp[k * 32 + lane];
    xv[k] = v; s1 += v; s2 += v * v;
  }
#pragma unroll
  for (int off = 16; off >= 1; off >>= 1) {
    s1 += __shfl_xor(s1, off);
    s2 += __shfl_xor(s2, off);
  }
  const float mu   = s1 * (1.f / 256.f);
  const float var  = s2 * (1.f / 256.f) - mu * mu;
  const float rstd = rsqrtf(var + 1e-5f);
#pragma unroll
  for (int k = 0; k < 8; ++k) {
    const int c = k * 32 + lane;
    xtok[wave][c] = (xv[k] - mu) * rstd * norm_w[c] + norm_b[c];
  }
  __syncthreads();

  float pi = proj_b[2 * lane], pj = proj_b[2 * lane + 1];
#pragma unroll 8
  for (int c = 0; c < 256; ++c) {
    const float xc = xtok[wave][c];
    const v2bf wv = *(const v2bf*)(&Wp[c * 64 + 2 * lane]);
    pi += xc * (float)wv[0];
    pj += xc * (float)wv[1];
  }
  const int r = l * 32 + lane;
  Ibuf[r * 128 + m] = (__bf16)pi;
  Jt[r * 128 + m]   = (__bf16)pj;
}

// ---------------------------------------------------------------------------
// f32 -> bf16 convert (final_w, 128x1024).
// ---------------------------------------------------------------------------
__global__ __launch_bounds__(256) void opm_cvt_bf16(
    const float* __restrict__ src, __bf16* __restrict__ dst, int n) {
  const int i = blockIdx.x * 256 + threadIdx.x;
  if (i < n) dst[i] = (__bf16)src[i];
}

// ---------------------------------------------------------------------------
// Kernel 2: fused outer-product-mean + final projection.
// Grid: x = lj-block (8 blocks of 32 lj), y = li-pair (128). 256 thr / 8 waves.
//   GEMM1: S(2 x 32x1024) = A(32x128) x B(128x1024) per li; K=m; bf16 wmma.
//   GEMM2: OUT(2 x 32x128) = Sflat(32x1024) x Wf^T per li; bf16 wmma.
// LDS ~146KB/WG (<= 320KB WGP LDS).
// ---------------------------------------------------------------------------
__global__ __launch_bounds__(256) void opm_main(
    const __bf16* __restrict__ Ibuf, const __bf16* __restrict__ Jt,
    const __bf16* __restrict__ Wfbf, const float* __restrict__ final_b,
    float* __restrict__ out) {
  constexpr int IPAD = 136;   // padded row stride (bf16) for I tiles (+4 banks/row)
  constexpr int SPAD = 1032;  // padded row stride (bf16) for S tiles (+4 banks/row)

  __shared__ __bf16 I_lds[2 * 32 * IPAD];   // 2 x (I_li^T 32x128), ~17KB
  __shared__ __bf16 S_lds[2 * 32 * SPAD];   // 2 x (Sflat 32x1024), ~129KB

  const int lib  = blockIdx.y;          // li pair: li = lib*2 + {0,1}
  const int ljb  = blockIdx.x;          // block of 32 lj values
  const int tid  = threadIdx.x;
  const int wave = tid >> 5;
  const int lane = tid & 31;
  const int ln16 = lane & 15;
  const int hi   = lane >> 4;

  // ---- Phase 0: async-stage both I_li^T tiles (16KB contiguous) into LDS.
  {
    const __bf16* src = Ibuf + (size_t)(lib * 2) * (32 * 128);
#pragma unroll
    for (int it = 0; it < 4; ++it) {
      const int seg = tid + it * 256;        // 1024 x 16B segments
      const int liL = seg >> 8;              // which li of the pair
      const int row = (seg >> 3) & 31;       // x row
      const int s8  = seg & 7;               // 16-bf16 segment in row
      async_copy_b128(src + liL * 4096 + row * 128 + s8 * 16,
                      I_lds + (liL * 32 + row) * IPAD + s8 * 16);
    }
    wait_async_all();
  }
  __syncthreads();

  // ---- Phase 1: GEMM1 for both li. Hoist all A fragments (2 li x 2 xt x 4 kt).
  v16bf afr[4][4];
#pragma unroll
  for (int li = 0; li < 2; ++li)
#pragma unroll
    for (int xt = 0; xt < 2; ++xt)
#pragma unroll
      for (int kt = 0; kt < 4; ++kt)
        afr[li * 2 + xt][kt] =
            load_a_frag(I_lds + (li * 32 + xt * 16 + ln16) * IPAD, kt * 32, hi);

  const float invM = 1.0f / 128.0f;
  for (int ntg = 0; ntg < 8; ++ntg) {
    const int ncol = (wave * 8 + ntg) * 16 + ln16;   // 0..1023 = lj_local*32 + y
    const __bf16* brow = Jt + (size_t)(ljb * 1024 + ncol) * 128;
    __builtin_prefetch(brow + 1024 * 128, 0, 0);     // next ljb block -> L2

    v8f acc[4];
#pragma unroll
    for (int a = 0; a < 4; ++a)
#pragma unroll
      for (int v = 0; v < 8; ++v) acc[a][v] = 0.f;

#pragma unroll
    for (int kt = 0; kt < 4; ++kt) {
      const v16bf bfrag = *(const v16bf*)(brow + kt * 32 + hi * 16);
#pragma unroll
      for (int a = 0; a < 4; ++a) acc[a] = wmma_bf16(afr[a][kt], bfrag, acc[a]);
    }

    // Scatter D (row = hi*8+v, col = ncol) into Sflat[li][lj][x*32+y] as bf16.
    const int lj = ncol >> 5;
    const int y  = ncol & 31;
#pragma unroll
    for (int li = 0; li < 2; ++li)
#pragma unroll
      for (int xt = 0; xt < 2; ++xt)
#pragma unroll
        for (int v = 0; v < 8; ++v) {
          const int x = xt * 16 + hi * 8 + v;
          S_lds[(li * 32 + lj) * SPAD + x * 32 + y] =
              (__bf16)(acc[li * 2 + xt][v] * invM);
        }
  }
  __syncthreads();

  // ---- Phase 2: OUT(32 lj x 128 o) per li; each Wf B-fragment feeds 4 wmma.
  const int ocol = wave * 16 + ln16;
  const __bf16* wrow = Wfbf + (size_t)ocol * 1024;

  v8f acc2[4];
#pragma unroll
  for (int a = 0; a < 4; ++a)
#pragma unroll
    for (int v = 0; v < 8; ++v) acc2[a][v] = 0.f;

#pragma unroll 4
  for (int kt = 0; kt < 32; ++kt) {
    const int kb = kt * 32;
    const v16bf bfrag = *(const v16bf*)(wrow + kb + hi * 16);
#pragma unroll
    for (int li = 0; li < 2; ++li)
#pragma unroll
      for (int mt = 0; mt < 2; ++mt) {
        const v16bf afrag =
            load_a_frag(S_lds + (li * 32 + mt * 16 + ln16) * SPAD, kb, hi);
        acc2[li * 2 + mt] = wmma_bf16(afrag, bfrag, acc2[li * 2 + mt]);
      }
  }

  const float bias = final_b[ocol];
#pragma unroll
  for (int li = 0; li < 2; ++li)
#pragma unroll
    for (int mt = 0; mt < 2; ++mt)
#pragma unroll
      for (int v = 0; v < 8; ++v) {
        const int ljg = ljb * 32 + mt * 16 + hi * 8 + v;
        const int lig = lib * 2 + li;
        out[((size_t)lig * 256 + ljg) * 128 + ocol] = acc2[li * 2 + mt][v] + bias;
      }
}

// ---------------------------------------------------------------------------
// Launch. Inputs (setup_inputs order):
//  0 x1d (1,128,256,256) f32   1 norm_w (256)   2 norm_b (256)
//  3 proj_w (64,256)           4 proj_b (64)
//  5 final_w (128,1024)        6 final_b (128)
// Output: (1,256,256,128) f32.  Workspace: Ibuf 2MB | Jt 2MB | Wfbf 256KB.
// ---------------------------------------------------------------------------
extern "C" void kernel_launch(void* const* d_in, const int* in_sizes, int n_in,
                              void* d_out, int out_size, void* d_ws, size_t ws_size,
                              hipStream_t stream) {
  const float* x1d     = (const float*)d_in[0];
  const float* norm_w  = (const float*)d_in[1];
  const float* norm_b  = (const float*)d_in[2];
  const float* proj_w  = (const float*)d_in[3];
  const float* proj_b  = (const float*)d_in[4];
  const float* final_w = (const float*)d_in[5];
  const float* final_b = (const float*)d_in[6];
  float* out = (float*)d_out;

  char* ws = (char*)d_ws;
  __bf16* Ibuf = (__bf16*)(ws);                        // 256*32*128 bf16 = 2MB
  __bf16* Jt   = (__bf16*)(ws + (2u << 20));           // 2MB
  __bf16* Wfbf = (__bf16*)(ws + (4u << 20));           // 128*1024 bf16 = 256KB

  opm_ln_proj<<<4096, 256, 0, stream>>>(x1d, norm_w, norm_b, proj_w, proj_b,
                                        Ibuf, Jt);
  opm_cvt_bf16<<<(128 * 1024 + 255) / 256, 256, 0, stream>>>(final_w, Wfbf,
                                                             128 * 1024);
  dim3 grid(8, 128);  // (lj-block of 32, li-pair)
  opm_main<<<grid, 256, 0, stream>>>(Ibuf, Jt, Wfbf, final_b, out);
}
